// Encoder_23527830848105
// MI455X (gfx1250) — matrix operand order repaired and checked by reading, not compiled, not run on hardware
//
#include <hip/hip_runtime.h>
#include <hip/hip_bf16.h>

// ---------------------------------------------------------------------------
// Transformer encoder (6 layers) for MI455X / gfx1250.
// - All GEMMs + attention matmuls: v_wmma_f32_16x16x32_bf16 (wave32 WMMA).
// - GEMM tiles staged with CDNA5 async global->LDS copies (ASYNCcnt),
//   double-buffered: fetch tile t+1 while WMMA consumes tile t.
// ---------------------------------------------------------------------------

typedef __bf16 v16bf __attribute__((ext_vector_type(16)));
typedef float  v8f   __attribute__((ext_vector_type(8)));

// Problem constants (fixed by the reference).
constexpr int CB  = 16;    // batch
constexpr int CS  = 512;   // sequence
constexpr int CD  = 512;   // model dim
constexpr int CH  = 8;     // heads
constexpr int CDh = 64;    // head dim
constexpr int CF  = 2048;  // FFN dim
constexpr int CL  = 6;     // layers

__device__ __forceinline__ v8f v8f_zero() {
    v8f z;
#pragma unroll
    for (int j = 0; j < 8; ++j) z[j] = 0.0f;
    return z;
}

__device__ __forceinline__ v8f wmma_bf16(v16bf a, v16bf b, v8f c) {
    return __builtin_amdgcn_wmma_f32_16x16x32_bf16(
        /*neg_a=*/false, a, /*neg_b=*/false, b,
        /*c_mod=*/(short)0, c, /*reuse_a=*/false, /*reuse_b=*/false);
}

// AS3->generic casts place the LDS byte offset in the low 32 bits of the
// generic pointer (shared aperture only occupies the high half).
__device__ __forceinline__ unsigned lds_off32(const void* p) {
    return (unsigned)(unsigned long long)p;
}

// CDNA5 async copy: 16 bytes per lane, global -> LDS, tracked by ASYNCcnt.
__device__ __forceinline__ void async_copy_b128(unsigned lds_addr,
                                                const void* gaddr) {
    asm volatile("global_load_async_to_lds_b128 %0, %1, off"
                 :: "v"(lds_addr), "v"(gaddr)
                 : "memory");
}

// Faithful torch-style .view(B*H, S, Dh) reinterpretation of a [B,S,H*Dh]
// row-major tensor. Flat offset of logical (b, head hh, seq s, dim d).
__device__ __forceinline__ size_t qkv_off(int b, int hh, int s, int d) {
    int ss = hh * 64 + (s >> 3);          // S/H = 64, (H*Dh)/Dh collapse
    int e  = ((s & 7) << 6) + d;          // position within H*Dh row
    return ((size_t)(b * CS + ss)) * (size_t)(CH * CDh) + (size_t)e;
}

// ---------------------------------------------------------------------------
// Embedding + sinusoidal PE (full-index formula, truncated at batch_length).
// ---------------------------------------------------------------------------
__global__ void embed_pe_kernel(const int* __restrict__ x,
                                const int* __restrict__ blen,
                                const float* __restrict__ emb,
                                float* __restrict__ h) {
    size_t idx = (size_t)blockIdx.x * blockDim.x + threadIdx.x;
    size_t total = (size_t)CB * CS * CD;
    if (idx >= total) return;
    int d = (int)(idx % CD);
    size_t bs = idx / CD;
    int s = (int)(bs % CS);
    int b = (int)(bs / CS);
    int tok = x[b * CS + s];
    float ang = (float)s * __powf(10000.0f, -(2.0f * (float)d) / (float)CD);
    float pe = (d & 1) ? __cosf(ang) : __sinf(ang);
    float m = (s < blen[b]) ? 1.0f : 0.0f;
    h[idx] = emb[(size_t)tok * CD + d] + pe * m;
}

// ---------------------------------------------------------------------------
// fp32 -> bf16 cast.
// ---------------------------------------------------------------------------
__global__ void cast_f32_bf16_kernel(const float* __restrict__ in,
                                     __bf16* __restrict__ out, size_t n) {
    size_t idx = (size_t)blockIdx.x * blockDim.x + threadIdx.x;
    if (idx < n) out[idx] = (__bf16)in[idx];
}

// ---------------------------------------------------------------------------
// Tiled bf16 WMMA GEMM: C[M,N] = A[M,K] @ B[K,N] (+bias) (ReLU?).
// 128 threads = 4 waves; workgroup tile 64x64; K-step 32 (one WMMA K).
// Double-buffered async global->LDS staging:
//   per tile each wave issues 4 async b128 copies (A: 2, B: 2);
//   while computing tile t, tile t+1's copies are in flight;
//   s_wait_asynccnt 4 retires tile t (async loads complete in order).
// A staged row-major (contiguous 16B fragment reads); B staged row-major
// (fragment gathered with conflict-free strided 2B LDS reads).
// ---------------------------------------------------------------------------
#define TILE_M 64
#define TILE_N 64
#define TILE_K 32
#define KP 40   // padded A row (bf16 elems); 80B row stride (16B multiple)
#define NP 72   // padded B row (bf16 elems); 144B row stride (16B multiple)

template <bool RELU, bool WF32, bool WBF16>
__global__ __launch_bounds__(128) void gemm_bf16_wmma(
    const __bf16* __restrict__ A, const __bf16* __restrict__ Bw,
    const float* __restrict__ bias, float* __restrict__ Cf,
    __bf16* __restrict__ Cb, int M, int N, int K) {
    __shared__ alignas(16) __bf16 As[2][TILE_M][KP];
    __shared__ alignas(16) __bf16 Bs[2][TILE_K][NP];   // row-major [k][n]

    const int t    = threadIdx.x;    // 0..127
    const int wave = t >> 5;         // 0..3 -> M sub-tile
    const int lane = t & 31;
    const int m0 = blockIdx.y * TILE_M;
    const int n0 = blockIdx.x * TILE_N;

    v8f acc[4];
#pragma unroll
    for (int i = 0; i < 4; ++i) acc[i] = v8f_zero();

    // Async stagers: 8 bf16 (16B) per lane per copy; 2 copies per array.
    const int raA = t >> 2;            // A row (0..31), +32 on pass 1
    const int caA = (t & 3) * 8;       // A col chunk
    const int rbB = t >> 3;            // B row (0..15), +16 on pass 1
    const int cbB = (t & 7) * 8;       // B col chunk

    // Fragment geometry (16-bit WMMA operand layout, wave32).
    const int mfr  = lane & 15;            // A row / C col selector
    const int ko   = (lane >> 4) * 8;      // unused; A fragment K base is kb16
    const int kb16 = (lane >> 4) * 16;     // B K base per lane half

    auto stage = [&](int buf, int k0) {
#pragma unroll
        for (int p = 0; p < 2; ++p) {
            int ra = raA + p * 32;
            async_copy_b128(lds_off32(&As[buf][ra][caA]),
                            A + (size_t)(m0 + ra) * K + k0 + caA);
            int rb = rbB + p * 16;
            async_copy_b128(lds_off32(&Bs[buf][rb][cbB]),
                            Bw + (size_t)(k0 + rb) * N + n0 + cbB);
        }
    };

    const int nk = K / TILE_K;
    stage(0, 0);

    for (int kt = 0; kt < nk; ++kt) {
        const int buf = kt & 1;
        if (kt + 1 < nk) {
            stage(buf ^ 1, (kt + 1) * TILE_K);
            asm volatile("s_wait_asynccnt 4" ::: "memory");
        } else {
            asm volatile("s_wait_asynccnt 0" ::: "memory");
        }
        __syncthreads();   // whole tile resident for all waves

        // A fragment: rows wave*16 .. wave*16+15, K-slice of 32.
        v16bf afr;
        const __bf16* arow = &As[buf][wave * 16 + mfr][0];
#pragma unroll
        for (int j = 0; j < 8; ++j) {
            afr[j]     = arow[kb16 + j];
            afr[j + 8] = arow[kb16 + 8 + j];
        }
#pragma unroll
        for (int nt = 0; nt < 4; ++nt) {
            v16bf bfr;
#pragma unroll
            for (int j = 0; j < 16; ++j)
                bfr[j] = Bs[buf][kb16 + j][nt * 16 + mfr];
            acc[nt] = wmma_bf16(afr, bfr, acc[nt]);
        }
        __syncthreads();   // protect buf from being re-staged next iter
    }

    // Epilogue: C layout — VGPR j holds row (j + 8*(lane>>4)), col = lane&15.
    const int rsel = (lane >> 4) * 8;
    const int col0 = n0 + (lane & 15);
#pragma unroll
    for (int nt = 0; nt < 4; ++nt) {
        int col = col0 + nt * 16;
        float bv = bias ? bias[col] : 0.0f;
#pragma unroll
        for (int j = 0; j < 8; ++j) {
            int row = m0 + wave * 16 + rsel + j;
            float val = acc[nt][j] + bv;
            if (RELU) val = fmaxf(val, 0.0f);
            if (WF32)  Cf[(size_t)row * N + col] = val;
            if (WBF16) Cb[(size_t)row * N + col] = (__bf16)val;
        }
    }
}

// ---------------------------------------------------------------------------
// Fused attention: one wave handles one (batch*head, 16-query tile).
// Scores (16x512 fp32) live in LDS; softmax via wave32 shuffles; both
// matmuls (Q·K^T and P·V) run on WMMA bf16.
// ---------------------------------------------------------------------------
__global__ __launch_bounds__(32) void attention_kernel(
    const __bf16* __restrict__ q, const __bf16* __restrict__ k,
    const __bf16* __restrict__ vv, const int* __restrict__ x,
    __bf16* __restrict__ ctx) {
    __shared__ float sc[16][CS + 8];

    const int lane = threadIdx.x;
    const int qt = blockIdx.x;       // query tile 0..31
    const int bh = blockIdx.y;       // 0..B*H-1
    const int b  = bh / CH;
    const int hh = bh % CH;
    const int mb = bh % CB;          // faithful torch .repeat(H,1,1) mask batch

    const int mfr  = lane & 15;
    const int ko   = (lane >> 4) * 8;
    const int kb16 = (lane >> 4) * 16;
    const int s_q  = qt * 16 + mfr;

    // Q fragments for dims [0,32) and [32,64).
    v16bf aq0, aq1;
    {
        const __bf16* qrow = q + qkv_off(b, hh, s_q, 0);
#pragma unroll
        for (int j = 0; j < 8; ++j) {
            aq0[j]     = qrow[kb16 + j];
            aq0[j + 8] = qrow[kb16 + 8 + j];
            aq1[j]     = qrow[32 + kb16 + j];
            aq1[j + 8] = qrow[32 + kb16 + 8 + j];
        }
    }

    // ---- scores = scale * Q K^T, masked, into LDS ----
    for (int kt = 0; kt < CS / 16; ++kt) {
        int key = kt * 16 + mfr;     // this lane's key column
        const __bf16* krow = k + qkv_off(b, hh, key, 0);
        v16bf bk0, bk1;
#pragma unroll
        for (int j = 0; j < 16; ++j) {
            bk0[j] = krow[kb16 + j];        // dims [0,32)
            bk1[j] = krow[32 + kb16 + j];   // dims [32,64)
        }
        v8f s8 = v8f_zero();
        s8 = wmma_bf16(aq0, bk0, s8);
        s8 = wmma_bf16(aq1, bk1, s8);

        int keyc = kt * 16 + (lane & 15);
        bool msk = (x[mb * CS + keyc] == 0);
#pragma unroll
        for (int j = 0; j < 8; ++j) {
            int r = j + ((lane >> 4) * 8);
            sc[r][keyc] = msk ? -1.0e9f : s8[j] * 0.125f;   // 1/sqrt(64)
        }
    }
    __syncthreads();

    // ---- softmax over keys (512) per row, wave32 shuffle reductions ----
    for (int r = 0; r < 16; ++r) {
        float mx = -3.0e38f;
#pragma unroll
        for (int i = 0; i < CS / 32; ++i) mx = fmaxf(mx, sc[r][lane + i * 32]);
#pragma unroll
        for (int off = 16; off > 0; off >>= 1) mx = fmaxf(mx, __shfl_xor(mx, off, 32));
        float sum = 0.0f;
#pragma unroll
        for (int i = 0; i < CS / 32; ++i) {
            float e = __expf(sc[r][lane + i * 32] - mx);
            sc[r][lane + i * 32] = e;
            sum += e;
        }
#pragma unroll
        for (int off = 16; off > 0; off >>= 1) sum += __shfl_xor(sum, off, 32);
        float inv = 1.0f / sum;
#pragma unroll
        for (int i = 0; i < CS / 32; ++i) sc[r][lane + i * 32] *= inv;
    }
    __syncthreads();

    // ---- ctx tile = P @ V  (K = 512 in steps of 32, N = 64 as 4 frags) ----
    v8f oacc[4];
#pragma unroll
    for (int i = 0; i < 4; ++i) oacc[i] = v8f_zero();

    for (int kt = 0; kt < CS / 32; ++kt) {
        v16bf ap;
        const float* prow = &sc[mfr][kt * 32];
#pragma unroll
        for (int j = 0; j < 8; ++j) {
            ap[j]     = (__bf16)prow[kb16 + j];
            ap[j + 8] = (__bf16)prow[kb16 + 8 + j];
        }
#pragma unroll
        for (int nt = 0; nt < 4; ++nt) {
            v16bf bvf;
#pragma unroll
            for (int j = 0; j < 16; ++j) {
                int key = kt * 32 + kb16 + j;
                bvf[j] = vv[qkv_off(b, hh, key, nt * 16 + mfr)];
            }
            oacc[nt] = wmma_bf16(ap, bvf, oacc[nt]);
        }
    }

#pragma unroll
    for (int nt = 0; nt < 4; ++nt) {
#pragma unroll
        for (int j = 0; j < 8; ++j) {
            int r  = j + ((lane >> 4) * 8);
            int sq = qt * 16 + r;
            ctx[qkv_off(b, hh, sq, nt * 16 + (lane & 15))] = (__bf16)oacc[nt][j];
        }
    }
}

// ---------------------------------------------------------------------------
// Residual add + LayerNorm; writes fp32 h and bf16 copy for next GEMM.
// One block (128 threads) per row of D=512.
// ---------------------------------------------------------------------------
__global__ __launch_bounds__(128) void add_layernorm_kernel(
    const float* __restrict__ hin, const float* __restrict__ dlt,
    const float* __restrict__ g, const float* __restrict__ bta,
    float* __restrict__ hout, __bf16* __restrict__ hb) {
    __shared__ float red[128];
    const int row = blockIdx.x;
    const int t = threadIdx.x;
    const float* hr = hin + (size_t)row * CD;
    const float* dr = dlt + (size_t)row * CD;

    float loc[4];
    float s = 0.0f;
#pragma unroll
    for (int i = 0; i < 4; ++i) {
        loc[i] = hr[t + i * 128] + dr[t + i * 128];
        s += loc[i];
    }
    red[t] = s;
    __syncthreads();
    for (int off = 64; off > 0; off >>= 1) {
        if (t < off) red[t] += red[t + off];
        __syncthreads();
    }
    float mean = red[0] / (float)CD;
    __syncthreads();

    float s2 = 0.0f;
#pragma unroll
    for (int i = 0; i < 4; ++i) {
        float c = loc[i] - mean;
        s2 += c * c;
    }
    red[t] = s2;
    __syncthreads();
    for (int off = 64; off > 0; off >>= 1) {
        if (t < off) red[t] += red[t + off];
        __syncthreads();
    }
    float var = red[0] / (float)CD;
    float inv = rsqrtf(var + 1e-5f);

#pragma unroll
    for (int i = 0; i < 4; ++i) {
        int d = t + i * 128;
        float y = (loc[i] - mean) * inv * g[d] + bta[d];
        hout[(size_t)row * CD + d] = y;
        hb[(size_t)row * CD + d] = (__bf16)y;
    }
}

// ---------------------------------------------------------------------------
// Host orchestration.
// ---------------------------------------------------------------------------
extern "C" void kernel_launch(void* const* d_in, const int* in_sizes, int n_in,
                              void* d_out, int out_size, void* d_ws,
                              size_t ws_size, hipStream_t stream) {
    (void)in_sizes; (void)n_in; (void)out_size; (void)ws_size;

    const int*   x     = (const int*)d_in[0];
    const int*   blen  = (const int*)d_in[1];
    const float* emb   = (const float*)d_in[2];
    const float* Wq    = (const float*)d_in[3];
    const float* Wk    = (const float*)d_in[4];
    const float* Wv    = (const float*)d_in[5];
    const float* Wo    = (const float*)d_in[6];
    const float* bo    = (const float*)d_in[7];
    const float* ln1g  = (const float*)d_in[8];
    const float* ln1b  = (const float*)d_in[9];
    const float* W1    = (const float*)d_in[10];
    const float* b1    = (const float*)d_in[11];
    const float* W2    = (const float*)d_in[12];
    const float* b2    = (const float*)d_in[13];
    const float* ln2g  = (const float*)d_in[14];
    const float* ln2b  = (const float*)d_in[15];

    constexpr size_t NBS  = (size_t)CB * CS;        // 8192 rows
    constexpr size_t NBSD = NBS * CD;               // activations
    constexpr size_t NBSF = NBS * CF;
    constexpr size_t WDD  = (size_t)CL * CD * CD;   // per-family DxD weights
    constexpr size_t WDF  = (size_t)CL * CD * CF;

    char* ws = (char*)d_ws;
    size_t off = 0;
    auto carve = [&](size_t bytes) {
        void* p = ws + off;
        off += (bytes + 255) & ~(size_t)255;
        return p;
    };
    float*  h    = (float*)carve(NBSD * 4);
    __bf16* hb   = (__bf16*)carve(NBSD * 2);
    __bf16* qb   = (__bf16*)carve(NBSD * 2);
    __bf16* kb   = (__bf16*)carve(NBSD * 2);
    __bf16* vb   = (__bf16*)carve(NBSD * 2);
    __bf16* ctxb = (__bf16*)carve(NBSD * 2);
    float*  of32 = (float*)carve(NBSD * 4);
    __bf16* f1b  = (__bf16*)carve(NBSF * 2);
    __bf16* wqb  = (__bf16*)carve(WDD * 2);
    __bf16* wkb  = (__bf16*)carve(WDD * 2);
    __bf16* wvb  = (__bf16*)carve(WDD * 2);
    __bf16* wob  = (__bf16*)carve(WDD * 2);
    __bf16* w1b  = (__bf16*)carve(WDF * 2);
    __bf16* w2b  = (__bf16*)carve(WDF * 2);

    auto cgrid = [](size_t n) { return (unsigned)((n + 255) / 256); };

    // Embedding + PE, then bf16 copies of h and all weights.
    embed_pe_kernel<<<cgrid(NBSD), 256, 0, stream>>>(x, blen, emb, h);
    cast_f32_bf16_kernel<<<cgrid(NBSD), 256, 0, stream>>>(h, hb, NBSD);
    cast_f32_bf16_kernel<<<cgrid(WDD), 256, 0, stream>>>(Wq, wqb, WDD);
    cast_f32_bf16_kernel<<<cgrid(WDD), 256, 0, stream>>>(Wk, wkb, WDD);
    cast_f32_bf16_kernel<<<cgrid(WDD), 256, 0, stream>>>(Wv, wvb, WDD);
    cast_f32_bf16_kernel<<<cgrid(WDD), 256, 0, stream>>>(Wo, wob, WDD);
    cast_f32_bf16_kernel<<<cgrid(WDF), 256, 0, stream>>>(W1, w1b, WDF);
    cast_f32_bf16_kernel<<<cgrid(WDF), 256, 0, stream>>>(W2, w2b, WDF);

    const dim3 gemmDD((unsigned)(CD / TILE_N), (unsigned)(NBS / TILE_M));
    const dim3 gemmDF((unsigned)(CF / TILE_N), (unsigned)(NBS / TILE_M));
    const dim3 attG((unsigned)(CS / 16), (unsigned)(CB * CH));

    for (int l = 0; l < CL; ++l) {
        const __bf16* lwq = wqb + (size_t)l * CD * CD;
        const __bf16* lwk = wkb + (size_t)l * CD * CD;
        const __bf16* lwv = wvb + (size_t)l * CD * CD;
        const __bf16* lwo = wob + (size_t)l * CD * CD;
        const __bf16* lw1 = w1b + (size_t)l * CD * CF;
        const __bf16* lw2 = w2b + (size_t)l * CF * CD;

        // Q, K, V projections -> bf16.
        gemm_bf16_wmma<false, false, true><<<gemmDD, 128, 0, stream>>>(
            hb, lwq, nullptr, nullptr, qb, (int)NBS, CD, CD);
        gemm_bf16_wmma<false, false, true><<<gemmDD, 128, 0, stream>>>(
            hb, lwk, nullptr, nullptr, kb, (int)NBS, CD, CD);
        gemm_bf16_wmma<false, false, true><<<gemmDD, 128, 0, stream>>>(
            hb, lwv, nullptr, nullptr, vb, (int)NBS, CD, CD);

        // Fused masked softmax attention.
        attention_kernel<<<attG, 32, 0, stream>>>(qb, kb, vb, x, ctxb);

        // Output projection (+bias) -> fp32.
        gemm_bf16_wmma<false, true, false><<<gemmDD, 128, 0, stream>>>(
            ctxb, lwo, bo + (size_t)l * CD, of32, nullptr, (int)NBS, CD, CD);

        // h = LN(h + o)
        add_layernorm_kernel<<<(unsigned)NBS, 128, 0, stream>>>(
            h, of32, ln1g + (size_t)l * CD, ln1b + (size_t)l * CD, h, hb);

        // FFN: relu(h W1 + b1) -> bf16; then W2 (+b2) -> fp32.
        gemm_bf16_wmma<true, false, true><<<gemmDF, 128, 0, stream>>>(
            hb, lw1, b1 + (size_t)l * CF, nullptr, f1b, (int)NBS, CF, CD);
        gemm_bf16_wmma<false, true, false><<<gemmDD, 128, 0, stream>>>(
            f1b, lw2, b2 + (size_t)l * CD, of32, nullptr, (int)NBS, CD, CF);

        // h = LN(h + f)
        add_layernorm_kernel<<<(unsigned)NBS, 128, 0, stream>>>(
            h, of32, ln2g + (size_t)l * CD, ln2b + (size_t)l * CD, h, hb);
    }

    hipMemcpyAsync(d_out, h, NBSD * sizeof(float), hipMemcpyDeviceToDevice,
                   stream);
}
